// MambaLayer_8521215115359
// MI455X (gfx1250) — matrix operand order repaired and checked by reading, not compiled, not run on hardware
//
#include <hip/hip_runtime.h>
#include <math.h>

// ---------------------------------------------------------------------------
// Mamba block (B=32, L=2048, d_model=512, d_inner=1024, N=16, dt_rank=32)
// for MI455X / gfx1250: all GEMMs via v_wmma_f32_16x16x32_bf16 (wave32),
// B-tiles staged with global_load_async_to_lds_b128 (ASYNCcnt), batched
// A-tile loads, selective scan as register-resident recurrence.
// ---------------------------------------------------------------------------

typedef __attribute__((ext_vector_type(16))) __bf16 bf16x16;
typedef __attribute__((ext_vector_type(8)))  float  f32x8;

union FragBF { bf16x16 v; uint4 q[2]; };

static constexpr int BB  = 32;
static constexpr int LL  = 2048;
static constexpr int DM  = 512;
static constexpr int DI  = 1024;
static constexpr int NST = 16;
static constexpr int RK  = 32;
static constexpr int BL  = BB * LL;      // 65536 rows

// GEMM tiling
static constexpr int BM = 128;
static constexpr int BN = 64;

enum { EPI_NONE = 0, EPI_BIAS = 1, EPI_GELU = 2, EPI_SOFTPLUS = 3 };

__device__ __forceinline__ float siluf(float x) {
    return x / (1.0f + __expf(-x));
}

// gfx1250 async global->LDS copy (16B per lane), tracked by ASYNCcnt
__device__ __forceinline__ void async_copy_b128(unsigned lds_off, const void* gaddr) {
    asm volatile("global_load_async_to_lds_b128 %0, %1, off"
                 :: "v"(lds_off), "v"(gaddr) : "memory");
}
__device__ __forceinline__ void wait_asynccnt0() {
    asm volatile("s_wait_asynccnt 0" ::: "memory");
}

// ---------------------------------------------------------------------------
// fp32 -> bf16 weight conversion (once per weight matrix)
// ---------------------------------------------------------------------------
__global__ __launch_bounds__(256) void cvt_f32_bf16(const float* __restrict__ s,
                                                    __bf16* __restrict__ d, int n) {
    int i = blockIdx.x * 256 + threadIdx.x;
    if (i < n) d[i] = (__bf16)s[i];
}

// ---------------------------------------------------------------------------
// C[M,N] = epi( A[M,K](f32) * W[N,K](bf16)^T + bias )   via WMMA bf16
// grid = (N/64, M/128), block = 256 (8 waves, 4x2 wave grid, 32x32 per wave)
// BK = 64 for K%64==0 GEMMs, BK = 32 for dt_proj (K=32)
// ---------------------------------------------------------------------------
template <int BK>
__global__ __launch_bounds__(256)
void gemm_bf16_wmma(const float* __restrict__ A, const __bf16* __restrict__ W,
                    float* __restrict__ C, const float* __restrict__ bias,
                    int M, int N, int K, int lda, int ldc, int mode) {
    constexpr int PITCH  = BK + 8;       // bank-friendly LDS row pitch (bf16)
    constexpr int ASTEPS = BK / 8;       // float4 loads per thread for A tile
    constexpr int ACOLS  = BK / 4;       // float4 segments per A row
    constexpr int BSTEPS = BK / 32;      // uint4 async copies per thread for B
    constexpr int BCOLS  = BK / 8;       // uint4 segments per B row

    __shared__ __align__(16) __bf16 sA[BM * PITCH];
    __shared__ __align__(16) __bf16 sB[BN * PITCH];

    const int tid  = threadIdx.x;
    const int lane = tid & 31;
    const int wave = tid >> 5;
    const int wm   = wave >> 1;          // 0..3 : wave row-tile
    const int wn   = wave & 1;           // 0..1 : wave col-tile
    const int row0 = blockIdx.y * BM;
    const int col0 = blockIdx.x * BN;

    const int l15  = lane & 15;
    const int lhi  = lane >> 4;          // 0 or 1

    f32x8 acc[2][2] = {};

    for (int k0 = 0; k0 < K; k0 += BK) {
        // ---- B tile: async copy straight into LDS (bf16, no conversion) ----
        #pragma unroll
        for (int i = 0; i < BSTEPS; ++i) {
            int linear = i * 256 + tid;
            int r   = linear / BCOLS;            // 0..63
            int seg = linear % BCOLS;
            async_copy_b128((unsigned)(unsigned long long)(uintptr_t)
                                &sB[r * PITCH + seg * 8],
                            W + (size_t)(col0 + r) * K + k0 + seg * 8);
        }
        // ---- A tile: batch all global loads, then convert+store ----
        float4 a4[ASTEPS];
        #pragma unroll
        for (int i = 0; i < ASTEPS; ++i) {
            int linear = i * 256 + tid;
            int r  = linear / ACOLS;             // 0..127
            int cs = linear % ACOLS;
            a4[i] = *(const float4*)(A + (size_t)(row0 + r) * lda + k0 + cs * 4);
        }
        #pragma unroll
        for (int i = 0; i < ASTEPS; ++i) {
            int linear = i * 256 + tid;
            int r  = linear / ACOLS;
            int cs = linear % ACOLS;
            __bf16* p = &sA[r * PITCH + cs * 4];
            p[0] = (__bf16)a4[i].x; p[1] = (__bf16)a4[i].y;
            p[2] = (__bf16)a4[i].z; p[3] = (__bf16)a4[i].w;
        }
        // prefetch next K tile (gfx1250 global_prefetch path)
        if (k0 + BK < K) {
            __builtin_prefetch(A + (size_t)(row0 + (tid >> 1)) * lda + k0 + BK, 0, 1);
            __builtin_prefetch(W + (size_t)(col0 + (tid >> 2)) * K + k0 + BK, 0, 1);
        }
        wait_asynccnt0();
        __syncthreads();

        // ---- compute: BK/32 WMMA K-steps, fragments per ISA 7.12.2 ----
        #pragma unroll
        for (int ks = 0; ks < BK; ks += 32) {
            FragBF aF[2], bF[2];
            const int b0 = ks + lhi * 8;         // A K-half select
            #pragma unroll
            for (int mi = 0; mi < 2; ++mi) {
                int r = wm * 32 + mi * 16 + l15;
                aF[mi].q[0] = *(const uint4*)&sA[r * PITCH + b0];
                aF[mi].q[1] = *(const uint4*)&sA[r * PITCH + b0 + 16];
            }
            const int kb = ks + lhi * 8;         // B K-half select
            #pragma unroll
            for (int ni = 0; ni < 2; ++ni) {
                int c = wn * 32 + ni * 16 + l15;
                bF[ni].q[0] = *(const uint4*)&sB[c * PITCH + kb];
                bF[ni].q[1] = *(const uint4*)&sB[c * PITCH + kb + 16];
            }
            #pragma unroll
            for (int mi = 0; mi < 2; ++mi)
                #pragma unroll
                for (int ni = 0; ni < 2; ++ni)
                    acc[mi][ni] = __builtin_amdgcn_wmma_f32_16x16x32_bf16(
                        false, aF[mi].v, false, bF[ni].v, (short)0, acc[mi][ni],
                        false, false);
        }
        __syncthreads();
    }

    // ---- epilogue: bias / GELU / softplus, scatter per C/D layout ----
    #pragma unroll
    for (int mi = 0; mi < 2; ++mi) {
        #pragma unroll
        for (int ni = 0; ni < 2; ++ni) {
            const int nn = col0 + wn * 32 + ni * 16 + l15;
            float bv = (mode != EPI_NONE) ? bias[nn] : 0.0f;
            #pragma unroll
            for (int j = 0; j < 8; ++j) {
                const int mm = row0 + wm * 32 + mi * 16 + j + (lhi << 3);
                float v = acc[mi][ni][j] + bv;
                if (mode == EPI_GELU)
                    v = 0.5f * v * (1.0f + erff(v * 0.70710678118654752f));
                else if (mode == EPI_SOFTPLUS)
                    v = (v > 20.0f) ? v : log1pf(__expf(v));
                C[(size_t)mm * ldc + nn] = v;
            }
        }
    }
}

// ---------------------------------------------------------------------------
// depthwise causal conv1d (4 taps) + SiLU : reads xi half of xz, writes u
// ---------------------------------------------------------------------------
__global__ __launch_bounds__(256)
void conv_silu(const float* __restrict__ xz, const float* __restrict__ cw,
               const float* __restrict__ cb, float* __restrict__ u) {
    int idx = blockIdx.x * 256 + threadIdx.x;   // over B*L*DI
    int d = idx & (DI - 1);
    int l = (idx >> 10) & (LL - 1);
    int b = idx >> 21;
    float acc = cb[d];
    #pragma unroll
    for (int k = 0; k < 4; ++k) {
        int ls = l - 3 + k;
        if (ls >= 0)
            acc += xz[((size_t)(b * LL + ls)) * (2 * DI) + d] * cw[d * 4 + k];
    }
    u[idx] = siluf(acc);
}

// ---------------------------------------------------------------------------
// selective scan: h[n] = h[n]*exp(dt*A[n]) + dt*u*B_t[n] ; y = sum h[n]*C_t[n]
// one thread per (b, d) channel; 16 states in registers; dy: delta in, y out
// ---------------------------------------------------------------------------
__global__ __launch_bounds__(128)
void scan_kernel(const float* __restrict__ u, float* __restrict__ dy,
                 const float* __restrict__ xdbl, const float* __restrict__ A_log) {
    const int d = blockIdx.y * 128 + threadIdx.x;   // 0..1023
    const int b = blockIdx.x;                       // 0..31
    float Ar[NST], h[NST];
    #pragma unroll
    for (int n = 0; n < NST; ++n) {
        Ar[n] = -__expf(A_log[d * NST + n]);
        h[n]  = 0.0f;
    }
    const size_t rowbase = (size_t)b * LL;
    for (int t = 0; t < LL; ++t) {
        const size_t r = rowbase + t;
        const float ut = u[r * DI + d];
        const float dt = dy[r * DI + d];
        const float* xb = xdbl + r * 64;
        const float du = dt * ut;
        float y = 0.0f;
        #pragma unroll
        for (int n = 0; n < NST; ++n) {
            const float dA = __expf(dt * Ar[n]);
            h[n] = h[n] * dA + du * xb[RK + n];        // B_t
            y += h[n] * xb[RK + NST + n];              // C_t
        }
        dy[r * DI + d] = y;
    }
}

// ---------------------------------------------------------------------------
// gate: y = (y + u*D) * silu(z)   (in-place on dy)
// ---------------------------------------------------------------------------
__global__ __launch_bounds__(256)
void gate_kernel(float* __restrict__ dy, const float* __restrict__ u,
                 const float* __restrict__ xz, const float* __restrict__ Dp) {
    int idx = blockIdx.x * 256 + threadIdx.x;   // over B*L*DI
    int d = idx & (DI - 1);
    size_t row = (size_t)(idx >> 10);
    float z = xz[row * (2 * DI) + DI + d];
    dy[idx] = (dy[idx] + u[idx] * Dp[d]) * siluf(z);
}

// ---------------------------------------------------------------------------
// residual add + LayerNorm(512), one wave32 per row (16 elems/lane, shfl_xor)
// ---------------------------------------------------------------------------
__global__ __launch_bounds__(256)
void add_ln(const float* __restrict__ x, const float* __restrict__ res,
            const float* __restrict__ g, const float* __restrict__ bb,
            float* __restrict__ out) {
    const int lane = threadIdx.x & 31;
    const int wv   = threadIdx.x >> 5;
    const size_t row = (size_t)blockIdx.x * 8 + wv;
    const float* xr = x + row * DM;
    const float* rr = res + row * DM;
    float v[16];
    float s = 0.0f;
    #pragma unroll
    for (int i = 0; i < 4; ++i) {
        int c = i * 128 + lane * 4;
        float4 a = *(const float4*)(xr + c);
        float4 b4 = *(const float4*)(rr + c);
        v[i * 4 + 0] = a.x + b4.x; v[i * 4 + 1] = a.y + b4.y;
        v[i * 4 + 2] = a.z + b4.z; v[i * 4 + 3] = a.w + b4.w;
        s += v[i * 4 + 0] + v[i * 4 + 1] + v[i * 4 + 2] + v[i * 4 + 3];
    }
    #pragma unroll
    for (int off = 16; off > 0; off >>= 1) s += __shfl_xor(s, off, 32);
    const float mu = s * (1.0f / 512.0f);
    float vs = 0.0f;
    #pragma unroll
    for (int i = 0; i < 16; ++i) { float dlt = v[i] - mu; vs += dlt * dlt; }
    #pragma unroll
    for (int off = 16; off > 0; off >>= 1) vs += __shfl_xor(vs, off, 32);
    const float inv = rsqrtf(vs * (1.0f / 512.0f) + 1e-12f);
    #pragma unroll
    for (int i = 0; i < 4; ++i) {
        int c = i * 128 + lane * 4;
        float4 o;
        o.x = (v[i * 4 + 0] - mu) * inv * g[c + 0] + bb[c + 0];
        o.y = (v[i * 4 + 1] - mu) * inv * g[c + 1] + bb[c + 1];
        o.z = (v[i * 4 + 2] - mu) * inv * g[c + 2] + bb[c + 2];
        o.w = (v[i * 4 + 3] - mu) * inv * g[c + 3] + bb[c + 3];
        *(float4*)(out + row * DM + c) = o;
    }
}

// ---------------------------------------------------------------------------
// Workspace layout (floats), with buffer aliasing:
//   xz [BL,2048]  -> later reused as f [BL,2048]
//   dy [BL,1024]  : delta -> y -> gated y
//   u  [BL,1024]  -> later reused as w2-out [BL,512]
//   xdbl [BL,64]
//   mo [BL,512]   : mamba-out -> h (LN1 in-place)
//   bf16 weights region
// ---------------------------------------------------------------------------
static constexpr size_t OFF_XZ = 0;
static constexpr size_t OFF_DY = OFF_XZ + (size_t)BL * 2 * DI;
static constexpr size_t OFF_U  = OFF_DY + (size_t)BL * DI;
static constexpr size_t OFF_XD = OFF_U  + (size_t)BL * DI;
static constexpr size_t OFF_MO = OFF_XD + (size_t)BL * 64;
static constexpr size_t OFF_WB = OFF_MO + (size_t)BL * DM;       // bf16 region

extern "C" void kernel_launch(void* const* d_in, const int* in_sizes, int n_in,
                              void* d_out, int out_size, void* d_ws, size_t ws_size,
                              hipStream_t stream) {
    const float* input    = (const float*)d_in[0];
    const float* in_proj  = (const float*)d_in[1];
    const float* conv_w   = (const float*)d_in[2];
    const float* conv_b   = (const float*)d_in[3];
    const float* x_proj   = (const float*)d_in[4];
    const float* dt_projw = (const float*)d_in[5];
    const float* dt_projb = (const float*)d_in[6];
    const float* A_log    = (const float*)d_in[7];
    const float* Dvec     = (const float*)d_in[8];
    const float* out_proj = (const float*)d_in[9];
    const float* ln1_g    = (const float*)d_in[10];
    const float* ln1_b    = (const float*)d_in[11];
    const float* w1_w     = (const float*)d_in[12];
    const float* w1_b     = (const float*)d_in[13];
    const float* w2_w     = (const float*)d_in[14];
    const float* w2_b     = (const float*)d_in[15];
    const float* ln2_g    = (const float*)d_in[16];
    const float* ln2_b    = (const float*)d_in[17];

    float* ws   = (float*)d_ws;
    float* xz   = ws + OFF_XZ;       // then f
    float* dy   = ws + OFF_DY;
    float* u    = ws + OFF_U;        // then w2-out
    float* xdbl = ws + OFF_XD;
    float* mo   = ws + OFF_MO;       // then h
    float* fbuf = xz;
    float* w2o  = u;
    float* outp = (float*)d_out;

    __bf16* wb = (__bf16*)(ws + OFF_WB);
    __bf16* wb_inproj  = wb;                         // 2048*512
    __bf16* wb_xproj   = wb_inproj  + 2048 * 512;    // 64*1024
    __bf16* wb_dtproj  = wb_xproj   + 64 * 1024;     // 1024*32
    __bf16* wb_outproj = wb_dtproj  + 1024 * 32;     // 512*1024
    __bf16* wb_w1      = wb_outproj + 512 * 1024;    // 2048*512
    __bf16* wb_w2      = wb_w1      + 2048 * 512;    // 512*2048

    // ---- weight conversions to bf16 ----
    auto cvt = [&](const float* s, __bf16* d, int n) {
        cvt_f32_bf16<<<(n + 255) / 256, 256, 0, stream>>>(s, d, n);
    };
    cvt(in_proj,  wb_inproj,  2048 * 512);
    cvt(x_proj,   wb_xproj,   64 * 1024);
    cvt(dt_projw, wb_dtproj,  1024 * 32);
    cvt(out_proj, wb_outproj, 512 * 1024);
    cvt(w1_w,     wb_w1,      2048 * 512);
    cvt(w2_w,     wb_w2,      512 * 2048);

    const dim3 blk(256);

    // 1) in_proj: xz[BL,2048] = input[BL,512] @ in_proj^T
    gemm_bf16_wmma<64><<<dim3(2048 / BN, BL / BM), blk, 0, stream>>>(
        input, wb_inproj, xz, nullptr, BL, 2048, 512, 512, 2048, EPI_NONE);

    // 2) depthwise conv + SiLU -> u
    conv_silu<<<(BL * DI) / 256, blk, 0, stream>>>(xz, conv_w, conv_b, u);

    // 3) x_proj: xdbl[BL,64] = u @ x_proj^T
    gemm_bf16_wmma<64><<<dim3(64 / BN, BL / BM), blk, 0, stream>>>(
        u, wb_xproj, xdbl, nullptr, BL, 64, 1024, 1024, 64, EPI_NONE);

    // 4) dt_proj + softplus: dy[BL,1024] = softplus(xdbl[:, :32] @ dt_proj^T + b)
    gemm_bf16_wmma<32><<<dim3(1024 / BN, BL / BM), blk, 0, stream>>>(
        xdbl, wb_dtproj, dy, dt_projb, BL, 1024, 32, 64, 1024, EPI_SOFTPLUS);

    // 5) selective scan (delta in dy -> y in dy)
    scan_kernel<<<dim3(BB, DI / 128), dim3(128), 0, stream>>>(u, dy, xdbl, A_log);

    // 6) gate: dy = (dy + u*D) * silu(z)
    gate_kernel<<<(BL * DI) / 256, blk, 0, stream>>>(dy, u, xz, Dvec);

    // 7) out_proj: mo[BL,512] = dy @ out_proj^T
    gemm_bf16_wmma<64><<<dim3(512 / BN, BL / BM), blk, 0, stream>>>(
        dy, wb_outproj, mo, nullptr, BL, 512, 1024, 1024, 512, EPI_NONE);

    // 8) h = LN1(mo + input)   (in-place on mo)
    add_ln<<<BL / 8, blk, 0, stream>>>(mo, input, ln1_g, ln1_b, mo);

    // 9) f = gelu(h @ w1^T + b1)   (f reuses xz buffer)
    gemm_bf16_wmma<64><<<dim3(2048 / BN, BL / BM), blk, 0, stream>>>(
        mo, wb_w1, fbuf, w1_b, BL, 2048, 512, 512, 2048, EPI_GELU);

    // 10) w2: w2o[BL,512] = f @ w2^T + b2   (w2o reuses u buffer)
    gemm_bf16_wmma<64><<<dim3(512 / BN, BL / BM), blk, 0, stream>>>(
        fbuf, wb_w2, w2o, w2_b, BL, 512, 2048, 2048, 512, EPI_BIAS);

    // 11) out = LN2(w2o + h)
    add_ln<<<BL / 8, blk, 0, stream>>>(w2o, mo, ln2_g, ln2_b, outp);

    (void)in_sizes; (void)n_in; (void)out_size; (void)ws_size;
}
